// MultiHeadAttention_52140902973776
// MI455X (gfx1250) — compile-verified, matrix operand order checked by reading, not run on hardware
//
#include <hip/hip_runtime.h>
#include <hip/hip_bf16.h>

// ---------------------------------------------------------------------------
// MultiHeadAttention forward for MI455X (gfx1250, wave32).
//   - all GEMMs: v_wmma_f32_16x16x32_bf16, double-buffered LDS tiles
//   - A tiles (bf16 sources): one TENSOR_LOAD_TO_LDS (TDM) per tile from wave 0,
//     descriptor pad feature reproduces the 40-bf16 LDS row stride
//   - B tiles: per-lane global_load_async_to_lds_b128 (GVS form, SGPR base)
//   - CONTEXT A tile (f32 softmax probs): VALU convert + ds_store_b128
//   pipeline per K-step: wait(ASYNCcnt[,TENSORcnt]) -> barrier -> issue next
//   tile's DMA -> WMMA current tile.
// ---------------------------------------------------------------------------

typedef __bf16 bf16_t;
typedef __attribute__((ext_vector_type(16))) __bf16 v16bf;
typedef __attribute__((ext_vector_type(8)))  __bf16 bf16x8;
typedef __attribute__((ext_vector_type(8)))  float  v8f;
typedef __attribute__((ext_vector_type(4)))  unsigned int u32x4;
typedef __attribute__((ext_vector_type(8)))  int          i32x8;
typedef __attribute__((ext_vector_type(4)))  int          i32x4;

#define B_  2
#define S_  2048
#define D_  1024
#define H_  16
#define DK_ 64

constexpr int BM = 128;
constexpr int BN = 64;
constexpr int BK = 32;
constexpr int LDS_STRIDE = BK + 8;   // 40 bf16 = 64B data + 16B pad per row

enum { MODE_PROJ = 0, MODE_PROJV = 1, MODE_SCORES = 2, MODE_CONTEXT = 3, MODE_OUT = 4 };

union Frag { v16bf v; bf16x8 h[2]; };

#if defined(__has_builtin)
#  if __has_builtin(__builtin_amdgcn_tensor_load_to_lds)
#    define HAVE_TDM 1
#  endif
#endif
#ifndef HAVE_TDM
#  define HAVE_TDM 0
#endif

// Per-lane async DMA of 16B global -> LDS, GVS form (SGPR base + u32 offset).
__device__ __forceinline__ void async_b128_gvs(const bf16_t* lds_dst,
                                               const bf16_t* base, unsigned off_bytes)
{
    unsigned int loff = (unsigned int)(unsigned long long)lds_dst; // low 32b = LDS offset
    asm volatile("global_load_async_to_lds_b128 %0, %1, %2"
                 :: "v"(loff), "v"(off_bytes), "s"(base) : "memory");
}
__device__ __forceinline__ void wait_async0()
{
    asm volatile("s_wait_asynccnt 0x0" ::: "memory");
}

#if HAVE_TDM
// One TDM load of a BM x BK bf16 tile (row stride = stride_elems) into LDS,
// with 16B padding inserted after every 64B row -> LDS stride 40 bf16.
__device__ __forceinline__ void tdm_load_tile_a(unsigned lds_addr,
                                                const bf16_t* gsrc,
                                                unsigned stride_elems)
{
    unsigned long long ga = (unsigned long long)gsrc;
    u32x4 g0;
    g0[0] = 1u;                                                  // count=1 (user D#)
    g0[1] = lds_addr;                                            // LDS byte address
    g0[2] = (unsigned)ga;                                        // global_addr[31:0]
    g0[3] = (unsigned)((ga >> 32) & 0x01FFFFFFu) | (2u << 30);   // addr[56:32] | type=2

    i32x8 g1;
    // data_size=1 (2B) | pad_enable | pad_interval=3 (16 DW = 64B) | pad_amount=3 (4 DW = 16B)
    g1[0] = (1 << 16) | (1 << 20) | (3 << 22) | (3 << 25);
    g1[1] = (int)((BK & 0xFFFF) << 16);                 // tensor_dim0[15:0] = BK
    g1[2] = (int)((BM & 0xFFFF) << 16);                 // td0 hi16=0 | tensor_dim1[15:0]=BM
    g1[3] = (int)((BK & 0xFFFF) << 16);                 // td1 hi16=0 | tile_dim0 = BK
    g1[4] = (int)(BM & 0xFFFF);                         // tile_dim1 = BM, tile_dim2 = 0
    g1[5] = (int)stride_elems;                          // tensor_dim0_stride[31:0]
    g1[6] = 0;                                          // stride hi | dim1_stride lo
    g1[7] = 0;

    i32x4 z4 = {0, 0, 0, 0};
#if defined(__clang_major__) && (__clang_major__ >= 23)
    i32x8 z8 = {0, 0, 0, 0, 0, 0, 0, 0};
    __builtin_amdgcn_tensor_load_to_lds(g0, g1, z4, z4, z8, 0);
#else
    __builtin_amdgcn_tensor_load_to_lds(g0, g1, z4, z4, 0);
#endif
}
#endif // HAVE_TDM

// ---------------------------------------------------------------------------
// Stage one BMxBK A tile and one BNxBK B(transposed) tile into LDS.
// ---------------------------------------------------------------------------
template <int MODE>
__device__ __forceinline__ void stage_tiles(
    int k0, int tid, int wave, int blockRow, int blockCol,
    bf16_t (* __restrict__ As)[LDS_STRIDE], bf16_t (* __restrict__ Bs)[LDS_STRIDE],
    const bf16_t* __restrict__ Ab, const float* __restrict__ Af,
    const bf16_t* __restrict__ Bb)
{
    constexpr int lda = (MODE == MODE_SCORES) ? DK_ : (MODE == MODE_CONTEXT ? S_ : D_);
    constexpr int ldb = lda;   // same leading dims per mode by construction

    // ---- A tile -------------------------------------------------------------
    if constexpr (MODE != MODE_CONTEXT) {
#if HAVE_TDM
        if (wave == 0)
            tdm_load_tile_a((unsigned)(unsigned long long)&As[0][0],
                            Ab + (size_t)blockRow * lda + k0, (unsigned)lda);
#else
        const int row  = tid >> 1;
        const int kseg = (tid & 1) << 4;
        const unsigned off =
            (unsigned)(((size_t)(blockRow + row) * lda + (k0 + kseg)) * 2);
        async_b128_gvs(&As[row][kseg + 0], Ab, off);
        async_b128_gvs(&As[row][kseg + 8], Ab, off + 16);
#endif
    } else {
        // f32 softmax probs -> convert to bf16 in registers, ds_store_b128
        const int row  = tid >> 1;
        const int kseg = (tid & 1) << 4;
        const float4* s4 =
            (const float4*)(Af + (size_t)(blockRow + row) * lda + (k0 + kseg));
        #pragma unroll
        for (int ii = 0; ii < 2; ++ii) {
            float4 v0 = s4[2 * ii + 0];
            float4 v1 = s4[2 * ii + 1];
            bf16x8 t;
            t[0] = (bf16_t)v0.x; t[1] = (bf16_t)v0.y;
            t[2] = (bf16_t)v0.z; t[3] = (bf16_t)v0.w;
            t[4] = (bf16_t)v1.x; t[5] = (bf16_t)v1.y;
            t[6] = (bf16_t)v1.z; t[7] = (bf16_t)v1.w;
            *(bf16x8*)&As[row][kseg + 8 * ii] = t;
        }
    }

    // ---- B tile (transposed, k-contiguous source) ---------------------------
    {
        const int n    = tid >> 2;
        const int kseg = (tid & 3) << 3;    // 0,8,16,24
        const unsigned off =
            (unsigned)(((size_t)(blockCol + n) * ldb + (k0 + kseg)) * 2);
        async_b128_gvs(&Bs[n][kseg], Bb, off);
    }
}

// ---------------------------------------------------------------------------
template <int MODE>
__global__ __launch_bounds__(256)
void gemm_wmma_kernel(const void* __restrict__ Ag, const void* __restrict__ Bg,
                      void* __restrict__ Cg, const float* __restrict__ bias)
{
    constexpr int K     = (MODE == MODE_SCORES) ? DK_ : (MODE == MODE_CONTEXT ? S_ : D_);
    constexpr int NSTEP = K / BK;
    constexpr bool kTdmA = HAVE_TDM && (MODE != MODE_CONTEXT);

    __shared__ __align__(16) bf16_t As[2][BM][LDS_STRIDE];   // double buffered
    __shared__ __align__(16) bf16_t Bs[2][BN][LDS_STRIDE];

    const int tid  = threadIdx.x;
    const int wave = tid >> 5;
    const int lane = tid & 31;
    const int half = lane >> 4;
    const int l16  = lane & 15;

    const int blockRow = blockIdx.x * BM;
    const int blockCol = blockIdx.y * BN;
    const int z        = blockIdx.z;        // (b*H + h) for SCORES/CONTEXT

    const bf16_t* Ab = nullptr; const float* Af = nullptr; const bf16_t* Bb = nullptr;
    if constexpr (MODE == MODE_PROJ || MODE == MODE_PROJV || MODE == MODE_OUT) {
        Ab = (const bf16_t*)Ag;                               // X / ctx [B*S, D] bf16
        Bb = (const bf16_t*)Bg;                               // W [D, D] bf16
    } else if constexpr (MODE == MODE_SCORES) {
        Ab = (const bf16_t*)Ag + (size_t)z * S_ * DK_;        // Q [S, DK]
        Bb = (const bf16_t*)Bg + (size_t)z * S_ * DK_;        // K [S, DK]
    } else {                                                  // CONTEXT
        Af = (const float*)Ag + (size_t)z * S_ * S_;          // P [S, S] f32
        Bb = (const bf16_t*)Bg + (size_t)z * DK_ * S_;        // Vt [DK, S] bf16
    }

    const int waveRow = (wave >> 1) * 32;   // 0,32,64,96
    const int waveCol = (wave & 1) * 32;    // 0,32

    v8f zero = {};
    v8f acc[2][2] = {{zero, zero}, {zero, zero}};

    // prologue fill of buffer 0
    stage_tiles<MODE>(0, tid, wave, blockRow, blockCol, As[0], Bs[0], Ab, Af, Bb);

    for (int kt = 0; kt < NSTEP; ++kt) {
        const int cur = kt & 1;

        wait_async0();                     // my async B fills (and A if fallback)
#if HAVE_TDM
        if constexpr (kTdmA) {
            if (wave == 0) __builtin_amdgcn_s_wait_tensorcnt(0);  // wave-0 TDM done
        }
#endif
        __syncthreads();                   // fills visible; prev compute retired

        if (kt + 1 < NSTEP)                // overlap next DMA with this tile's WMMA
            stage_tiles<MODE>((kt + 1) * BK, tid, wave, blockRow, blockCol,
                              As[cur ^ 1], Bs[cur ^ 1], Ab, Af, Bb);

        Frag fa[2], fb[2];
        #pragma unroll
        for (int i = 0; i < 2; ++i) {
            const int r = waveRow + i * 16 + l16;
            fa[i].h[0] = *(const bf16x8*)&As[cur][r][half * 8];       // K= 8h+0..7
            fa[i].h[1] = *(const bf16x8*)&As[cur][r][16 + half * 8];  // K=16+8h+0..7
        }
        #pragma unroll
        for (int j = 0; j < 2; ++j) {
            const int c = waveCol + j * 16 + l16;
            fb[j].h[0] = *(const bf16x8*)&Bs[cur][c][half * 8];
            fb[j].h[1] = *(const bf16x8*)&Bs[cur][c][16 + half * 8];
        }
        #pragma unroll
        for (int i = 0; i < 2; ++i)
            #pragma unroll
            for (int j = 0; j < 2; ++j)
                acc[i][j] = __builtin_amdgcn_wmma_f32_16x16x32_bf16(
                    false, fa[i].v, false, fb[j].v, (short)0, acc[i][j],
                    false, false);
    }

    // ---- epilogue: per-mode store -------------------------------------------
    #pragma unroll
    for (int i = 0; i < 2; ++i) {
        #pragma unroll
        for (int j = 0; j < 2; ++j) {
            const int tr = blockRow + waveRow + i * 16;
            const int tc = blockCol + waveCol + j * 16;
            #pragma unroll
            for (int r = 0; r < 8; ++r) {
                const int gr = tr + r + 8 * half;   // C: M = r + 8*(lane/16)
                const int gc = tc + l16;            //    N = lane%16
                float vv = acc[i][j][r];
                if constexpr (MODE == MODE_PROJ) {
                    const int b = gr / S_, s = gr % S_;
                    const int h = gc >> 6, dk = gc & 63;
                    ((bf16_t*)Cg)[((size_t)(b * H_ + h) * S_ + s) * DK_ + dk] =
                        (bf16_t)vv;
                } else if constexpr (MODE == MODE_PROJV) {      // store V transposed
                    const int b = gr / S_, s = gr % S_;
                    const int h = gc >> 6, dk = gc & 63;
                    ((bf16_t*)Cg)[((size_t)(b * H_ + h) * DK_ + dk) * S_ + s] =
                        (bf16_t)vv;
                } else if constexpr (MODE == MODE_SCORES) {
                    ((float*)Cg)[(size_t)z * S_ * S_ + (size_t)gr * S_ + gc] =
                        vv * 0.125f;   // 1/sqrt(DK)
                } else if constexpr (MODE == MODE_CONTEXT) {
                    const int b = z / H_, h = z % H_;
                    ((bf16_t*)Cg)[((size_t)(b * S_ + gr)) * D_ + h * DK_ + gc] =
                        (bf16_t)vv;
                } else {
                    ((float*)Cg)[(size_t)gr * D_ + gc] = vv + bias[gc];
                }
            }
        }
    }
}

// ---------------------------------------------------------------------------
// One-shot f32 -> bf16 conversion, 8 elements/thread (n multiple of 2048)
// ---------------------------------------------------------------------------
__global__ __launch_bounds__(256)
void cvt_f32_bf16_kernel(const float* __restrict__ src, bf16_t* __restrict__ dst)
{
    const size_t i = ((size_t)blockIdx.x * 256 + threadIdx.x) * 8;
    const float4* s4 = (const float4*)(src + i);
    float4 v0 = s4[0];
    float4 v1 = s4[1];
    bf16x8 t;
    t[0] = (bf16_t)v0.x; t[1] = (bf16_t)v0.y; t[2] = (bf16_t)v0.z; t[3] = (bf16_t)v0.w;
    t[4] = (bf16_t)v1.x; t[5] = (bf16_t)v1.y; t[6] = (bf16_t)v1.z; t[7] = (bf16_t)v1.w;
    *(bf16x8*)(dst + i) = t;
}

// ---------------------------------------------------------------------------
// In-place causal softmax over one row (b,h,q); zeros above the diagonal.
// ---------------------------------------------------------------------------
__global__ __launch_bounds__(256)
void softmax_causal_kernel(float* __restrict__ attn)
{
    const int   row = blockIdx.x;          // z*S + q
    const int   q   = row & (S_ - 1);
    float*      p   = attn + (size_t)row * S_;
    const int   len = q + 1;
    const int   tid = threadIdx.x;

    __shared__ float red[8];

    float m = -1e30f;
    for (int k = tid; k < len; k += 256) m = fmaxf(m, p[k]);
    #pragma unroll
    for (int o = 16; o > 0; o >>= 1) m = fmaxf(m, __shfl_down(m, o, 32));
    if ((tid & 31) == 0) red[tid >> 5] = m;
    __syncthreads();
    if (tid < 8) {
        float t = red[tid];
        #pragma unroll
        for (int o = 4; o > 0; o >>= 1) t = fmaxf(t, __shfl_down(t, o, 8));
        if (tid == 0) red[0] = t;
    }
    __syncthreads();
    m = red[0];
    __syncthreads();

    float s = 0.f;
    for (int k = tid; k < len; k += 256) s += __expf(p[k] - m);
    #pragma unroll
    for (int o = 16; o > 0; o >>= 1) s += __shfl_down(s, o, 32);
    if ((tid & 31) == 0) red[tid >> 5] = s;
    __syncthreads();
    if (tid < 8) {
        float t = red[tid];
        #pragma unroll
        for (int o = 4; o > 0; o >>= 1) t += __shfl_down(t, o, 8);
        if (tid == 0) red[0] = t;
    }
    __syncthreads();
    const float inv = 1.0f / red[0];

    for (int k = tid; k < len; k += 256) p[k] = __expf(p[k] - m) * inv;
    for (int k = len + tid; k < S_; k += 256) p[k] = 0.0f;
}

// ---------------------------------------------------------------------------
extern "C" void kernel_launch(void* const* d_in, const int* in_sizes, int n_in,
                              void* d_out, int out_size, void* d_ws, size_t ws_size,
                              hipStream_t stream)
{
    (void)in_sizes; (void)n_in; (void)out_size; (void)ws_size;

    const float* query = (const float*)d_in[0];
    const float* key   = (const float*)d_in[1];
    const float* value = (const float*)d_in[2];
    /* d_in[3] = mask: causal, handled analytically */
    const float* w_q   = (const float*)d_in[4];
    const float* w_k   = (const float*)d_in[5];
    const float* w_v   = (const float*)d_in[6];
    const float* w_o   = (const float*)d_in[7];
    const float* b_o   = (const float*)d_in[8];

    float* out  = (float*)d_out;
    float* attn = out + (size_t)B_ * S_ * D_;

    // workspace (bf16): Xq,Xk,Xv [B*S,D]; Wq..Wo [D,D]; Qb,Kb [B,H,S,DK];
    //                   Vt [B,H,DK,S]; Ctx [B,S,D]   (~67 MB total)
    const size_t nX = (size_t)B_ * S_ * D_;
    const size_t nW = (size_t)D_ * D_;
    bf16_t* Xq  = (bf16_t*)d_ws;
    bf16_t* Xk  = Xq + nX;
    bf16_t* Xv  = Xk + nX;
    bf16_t* Wq  = Xv + nX;
    bf16_t* Wk  = Wq + nW;
    bf16_t* Wv  = Wk + nW;
    bf16_t* Wo  = Wv + nW;
    bf16_t* Qb  = Wo + nW;
    bf16_t* Kb  = Qb + nX;
    bf16_t* Vt  = Kb + nX;
    bf16_t* Ctx = Vt + nX;

    dim3 blk(256);

    // 0) one-shot bf16 conversions (8 elems/thread)
    cvt_f32_bf16_kernel<<<nX / 2048, blk, 0, stream>>>(query, Xq);
    cvt_f32_bf16_kernel<<<nX / 2048, blk, 0, stream>>>(key,   Xk);
    cvt_f32_bf16_kernel<<<nX / 2048, blk, 0, stream>>>(value, Xv);
    cvt_f32_bf16_kernel<<<nW / 2048, blk, 0, stream>>>(w_q, Wq);
    cvt_f32_bf16_kernel<<<nW / 2048, blk, 0, stream>>>(w_k, Wk);
    cvt_f32_bf16_kernel<<<nW / 2048, blk, 0, stream>>>(w_v, Wv);
    cvt_f32_bf16_kernel<<<nW / 2048, blk, 0, stream>>>(w_o, Wo);

    // 1) projections
    dim3 gProj(B_ * S_ / BM, D_ / BN, 1);
    gemm_wmma_kernel<MODE_PROJ ><<<gProj, blk, 0, stream>>>(Xq, Wq, Qb, nullptr);
    gemm_wmma_kernel<MODE_PROJ ><<<gProj, blk, 0, stream>>>(Xk, Wk, Kb, nullptr);
    gemm_wmma_kernel<MODE_PROJV><<<gProj, blk, 0, stream>>>(Xv, Wv, Vt, nullptr);

    // 2) scores: per (b,h)  Q K^T / 8 -> attn region of d_out
    dim3 gScores(S_ / BM, S_ / BN, B_ * H_);
    gemm_wmma_kernel<MODE_SCORES><<<gScores, blk, 0, stream>>>(Qb, Kb, attn, nullptr);

    // 3) causal softmax in place
    softmax_causal_kernel<<<B_ * H_ * S_, blk, 0, stream>>>(attn);

    // 4) context: per (b,h)  P V -> bf16 ctx [B,S,D]
    dim3 gCtx(S_ / BM, DK_ / BN, B_ * H_);
    gemm_wmma_kernel<MODE_CONTEXT><<<gCtx, blk, 0, stream>>>(attn, Vt, Ctx, nullptr);

    // 5) output projection + bias -> f32 d_out
    dim3 gOut(B_ * S_ / BM, D_ / BN, 1);
    gemm_wmma_kernel<MODE_OUT><<<gOut, blk, 0, stream>>>(Ctx, Wo, out, b_o);
}
